// Attention_15496242004691
// MI455X (gfx1250) — matrix operand order repaired and checked by reading, not compiled, not run on hardware
//
#include <hip/hip_runtime.h>
#include <stdint.h>

typedef unsigned short     u16;
typedef unsigned int       u32;
typedef unsigned long long u64;

typedef __attribute__((ext_vector_type(16))) __bf16 v16bf;
typedef __attribute__((ext_vector_type(8)))  float  v8f;
typedef __attribute__((ext_vector_type(4)))  unsigned int v4u;
typedef __attribute__((ext_vector_type(8)))  int    v8i;
typedef __attribute__((ext_vector_type(4)))  int    v4i;

struct alignas(16) U4 { u32 x, y, z, w; };
union FragCvt { U4 q[2]; v16bf v; };

__device__ __forceinline__ v16bf mkfrag(U4 lo, U4 hi) {
  FragCvt f; f.q[0] = lo; f.q[1] = hi; return f.v;
}

// fp32 -> bf16 round-to-nearest-even (raw bits)
__device__ __forceinline__ u16 f2bf(float f) {
  u32 u = __float_as_uint(f);
  u32 r = (u + 0x7FFFu + ((u >> 16) & 1u)) >> 16;
  return (u16)r;
}

#define WMMA_BF16(a, b, c) \
  __builtin_amdgcn_wmma_f32_16x16x32_bf16(false, (a), false, (b), (short)0, (c), false, false)

// ---------------------------------------------------------------------------
// Tensor Data Mover: DMA a 2D tile (tw x th elements, 2B each, row stride
// `stride` elements) from global memory into LDS at byte offset lds_addr.
// D# built per CDNA5 ISA 08_async_tensor 8.3/8.4. f1 carries pad_enable/
// pad_interval/pad_amount bits for group1 dword0.
// ---------------------------------------------------------------------------
__device__ __forceinline__ void tdm_load_2d(u32 lds_addr, u64 gaddr,
                                            u32 tw, u32 th, u32 stride, u32 f1) {
  v4u g0 = { 1u,                                    // count=1 valid descriptor
             lds_addr,                              // LDS byte address
             (u32)gaddr,                            // global addr [31:0]
             (u32)((gaddr >> 32) & 0x1FFFFFFu) | (2u << 30) };  // addr[56:32] | type=2
  v8i g1 = { (int)((1u << 16) | f1),                // data_size=2B | pad cfg
             (int)(tw << 16),                       // tensor_dim0[15:0] @63:48
             (int)((tw >> 16) | (th << 16)),        // tensor_dim0 hi | tensor_dim1 lo
             (int)((th >> 16) | (tw << 16)),        // tensor_dim1 hi | tile_dim0
             (int)(th & 0xFFFFu),                   // tile_dim1 | tile_dim2=0
             (int)stride,                           // tensor_dim0_stride[31:0]
             0, 0 };
  v4i z4 = { 0, 0, 0, 0 };
#if defined(__clang_major__) && (__clang_major__ >= 23)
  v8i z8 = { 0, 0, 0, 0, 0, 0, 0, 0 };
  __builtin_amdgcn_tensor_load_to_lds(g0, g1, z4, z4, z8, 0);
#else
  __builtin_amdgcn_tensor_load_to_lds(g0, g1, z4, z4, 0);
#endif
}

#define BB    2
#define SS    2048
#define DIMM  2048
#define HH    16
#define DQK   192
#define DNOPE 128
#define DV    128
#define BSROWS (BB * SS)   // 4096

// pad 4 DWORDs (8 elems) after every 16 DWORDs (32 elems): 32-elem rows land
// on a 40-element LDS stride, matching the fragment-read addressing below.
#define TDM_PAD40 ((1u << 20) | (3u << 22) | (3u << 25))

// ---------------------------------------------------------------------------
// fp32 -> bf16 conversion (grid-stride)
// ---------------------------------------------------------------------------
__global__ __launch_bounds__(256) void cvt_f32_bf16(const float* __restrict__ in,
                                                    u16* __restrict__ out, int n) {
  for (int i = blockIdx.x * blockDim.x + threadIdx.x; i < n; i += gridDim.x * blockDim.x)
    out[i] = f2bf(in[i]);
}

// ---------------------------------------------------------------------------
// RMSNorm over C=512 cols (in_stride may be larger), bf16 output
// ---------------------------------------------------------------------------
__global__ __launch_bounds__(256) void rmsnorm_bf16(const float* __restrict__ in,
                                                    const float* __restrict__ w,
                                                    u16* __restrict__ out,
                                                    int in_stride, int C) {
  __shared__ float red[256];
  const int row = blockIdx.x;
  const float* src = in + (size_t)row * in_stride;
  float s = 0.f;
  for (int c = threadIdx.x; c < C; c += 256) { float v = src[c]; s += v * v; }
  red[threadIdx.x] = s;
  __syncthreads();
  for (int off = 128; off > 0; off >>= 1) {
    if (threadIdx.x < off) red[threadIdx.x] += red[threadIdx.x + off];
    __syncthreads();
  }
  const float rs = rsqrtf(red[0] / (float)C + 1e-6f);
  u16* dst = out + (size_t)row * C;
  for (int c = threadIdx.x; c < C; c += 256) dst[c] = f2bf(src[c] * rs * w[c]);
}

// ---------------------------------------------------------------------------
// bf16 GEMM, C[M,N] = A[M,K] * B[N,K]^T, fp32 accumulate via WMMA.
// Block: 256 thr (8 waves), tile 64x64; wave (4x2) -> 16x32 C sub-tile.
// K chunks of 32 staged by the Tensor Data Mover into double-buffered LDS
// (TDM pad feature produces the 40-element anti-conflict stride in HW).
// ---------------------------------------------------------------------------
__global__ __launch_bounds__(256) void gemm_bf16_nt(const u16* __restrict__ A,
                                                    const u16* __restrict__ Bm,
                                                    float* __restrict__ C,
                                                    int M, int N, int K) {
  __shared__ u16 sA[2][64 * 40];
  __shared__ u16 sB[2][64 * 40];
  const int tid  = threadIdx.x;
  const int lane = tid & 31, wave = tid >> 5;
  const int m = lane & 15, hs = lane >> 4;
  const int wm = wave & 3, wn = wave >> 2;
  const int mrow0 = blockIdx.y * 64;
  const int ncol0 = blockIdx.x * 64;
  const int mbase = mrow0 + wm * 16;
  const int nbase = ncol0 + wn * 32;
  v8f acc0 = {0.f,0.f,0.f,0.f,0.f,0.f,0.f,0.f};
  v8f acc1 = {0.f,0.f,0.f,0.f,0.f,0.f,0.f,0.f};

  const u32 ldsA0 = (u32)(uintptr_t)&sA[0][0], ldsA1 = (u32)(uintptr_t)&sA[1][0];
  const u32 ldsB0 = (u32)(uintptr_t)&sB[0][0], ldsB1 = (u32)(uintptr_t)&sB[1][0];
  const u64 baseA = (u64)(uintptr_t)(A  + (size_t)mrow0 * K);
  const u64 baseB = (u64)(uintptr_t)(Bm + (size_t)ncol0 * K);
  const int pr = tid >> 2, pc = (tid & 3) * 8;   // per-thread prefetch pattern

  if (wave == 0) {
    tdm_load_2d(ldsA0, baseA, 32, 64, (u32)K, TDM_PAD40);
    tdm_load_2d(ldsB0, baseB, 32, 64, (u32)K, TDM_PAD40);
    __builtin_amdgcn_s_wait_tensorcnt(0);
  }
  __syncthreads();

  int cur = 0;
  for (int k0 = 0; k0 < K; k0 += 32) {
    const int kn = k0 + 32;
    // kick DMA for the next chunk into the other buffer, overlapped with WMMA
    if (kn < K && wave == 0) {
      const u32 la = cur ? ldsA0 : ldsA1;
      const u32 lb = cur ? ldsB0 : ldsB1;
      tdm_load_2d(la, baseA + (u64)kn * 2, 32, 64, (u32)K, TDM_PAD40);
      tdm_load_2d(lb, baseB + (u64)kn * 2, 32, 64, (u32)K, TDM_PAD40);
    }
    // warm GL2 a few chunks ahead
    if (kn + 96 < K) {
      __builtin_prefetch((const char*)(uintptr_t)(baseA + (u64)(kn + 96) * 2) +
                         (size_t)pr * K * 2 + pc * 2, 0, 1);
      __builtin_prefetch((const char*)(uintptr_t)(baseB + (u64)(kn + 96) * 2) +
                         (size_t)pr * K * 2 + pc * 2, 0, 1);
    }

    const u16* cA = &sA[cur][0];
    const u16* cB = &sB[cur][0];
    // A frag: row m; elems e -> K = hs*16 + e (contiguous)
    U4 alo = *(const U4*)&cA[(wm * 16 + m) * 40 + hs * 16];
    U4 ahi = *(const U4*)&cA[(wm * 16 + m) * 40 + hs * 16 + 8];
    v16bf af = mkfrag(alo, ahi);
    // B frag: col = row of Bm tile; elems e -> K = hs*16 + e (contiguous)
    U4 b0lo = *(const U4*)&cB[(wn * 32 + m) * 40 + hs * 16];
    U4 b0hi = *(const U4*)&cB[(wn * 32 + m) * 40 + hs * 16 + 8];
    U4 b1lo = *(const U4*)&cB[(wn * 32 + 16 + m) * 40 + hs * 16];
    U4 b1hi = *(const U4*)&cB[(wn * 32 + 16 + m) * 40 + hs * 16 + 8];
    acc0 = WMMA_BF16(af, mkfrag(b0lo, b0hi), acc0);
    acc1 = WMMA_BF16(af, mkfrag(b1lo, b1hi), acc1);

    if (kn < K && wave == 0) __builtin_amdgcn_s_wait_tensorcnt(0);
    __syncthreads();
    cur ^= 1;
  }

#pragma unroll
  for (int r = 0; r < 8; ++r) {
    const int row = mbase + r + 8 * hs;   // D layout: row = vgpr + 8*half
    C[(size_t)row * N + nbase + m]      = acc0[r];
    C[(size_t)row * N + nbase + 16 + m] = acc1[r];
  }
}

// ---------------------------------------------------------------------------
// Q: copy nope + RoPE pe, fp32 (B*S, H*192) -> bf16 [B][H][S][192]
// ---------------------------------------------------------------------------
__global__ __launch_bounds__(96) void rope_q_kernel(const float* __restrict__ qf,
                                                    const float* __restrict__ fc,
                                                    u16* __restrict__ qbf) {
  const int bsh = blockIdx.x;
  const int h = bsh & 15, bs = bsh >> 4;
  const int s = bs & (SS - 1), b = bs >> 11;
  const float* src = qf + (size_t)bs * (HH * DQK) + h * DQK;
  u16* dst = qbf + (((size_t)b * HH + h) * SS + s) * DQK;
  const int t = threadIdx.x;
  if (t < 64) {
    dst[2 * t]     = f2bf(src[2 * t]);
    dst[2 * t + 1] = f2bf(src[2 * t + 1]);
  } else {
    const int i = t - 64;
    const float xr = src[DNOPE + 2 * i], xi = src[DNOPE + 2 * i + 1];
    const float c  = fc[(size_t)s * 64 + 2 * i], sn = fc[(size_t)s * 64 + 2 * i + 1];
    dst[DNOPE + 2 * i]     = f2bf(xr * c - xi * sn);
    dst[DNOPE + 2 * i + 1] = f2bf(xr * sn + xi * c);
  }
}

// ---------------------------------------------------------------------------
// K/V assembly: k_nope|rope(k_pe) -> k_bf [B][H][S][192], v -> v_bf [B][H][S][128]
// ---------------------------------------------------------------------------
__global__ __launch_bounds__(96) void assemble_kv_kernel(const float* __restrict__ kvb,
                                                         const float* __restrict__ kva,
                                                         const float* __restrict__ fc,
                                                         u16* __restrict__ kbf,
                                                         u16* __restrict__ vbf) {
  const int bsh = blockIdx.x;
  const int h = bsh & 15, bs = bsh >> 4;
  const int s = bs & (SS - 1), b = bs >> 11;
  const float* kn = kvb + (size_t)bs * (HH * 256) + h * 256;
  const float* vv = kn + DNOPE;
  const float* pe = kva + (size_t)bs * 576 + 512;
  u16* kd = kbf + (((size_t)b * HH + h) * SS + s) * DQK;
  u16* vd = vbf + (((size_t)b * HH + h) * SS + s) * DV;
  const int t = threadIdx.x;
  if (t < 64) {
    kd[2 * t]     = f2bf(kn[2 * t]);
    kd[2 * t + 1] = f2bf(kn[2 * t + 1]);
    vd[2 * t]     = f2bf(vv[2 * t]);
    vd[2 * t + 1] = f2bf(vv[2 * t + 1]);
  } else {
    const int i = t - 64;
    const float xr = pe[2 * i], xi = pe[2 * i + 1];
    const float c  = fc[(size_t)s * 64 + 2 * i], sn = fc[(size_t)s * 64 + 2 * i + 1];
    kd[DNOPE + 2 * i]     = f2bf(xr * c - xi * sn);
    kd[DNOPE + 2 * i + 1] = f2bf(xr * sn + xi * c);
  }
}

// ---------------------------------------------------------------------------
// Flash attention (causal), one (b,h) per blockIdx.y, 64 query rows per block.
// 4 waves; wave owns 16 query rows. 32-key tiles: K tile DMA'd by the TDM
// (overlapped with the cooperative transposed V staging), QK^T and PV via WMMA.
// ---------------------------------------------------------------------------
__global__ __launch_bounds__(128) void mla_flash_attn(const u16* __restrict__ qbf,
                                                      const u16* __restrict__ kbf,
                                                      const u16* __restrict__ vbf,
                                                      u16* __restrict__ attn) {
  __shared__ u16 sK[32 * 192];     // K tile [key][192] (TDM-filled, unpadded)
  __shared__ u16 sV[128 * 40];     // V tile transposed [col][key], padded stride
  __shared__ u16 sP[4][16 * 32];   // per-wave P relayout scratch
  const int bh = blockIdx.y;
  const int b = bh >> 4, head = bh & 15;
  const int qtile = blockIdx.x;
  const int tid = threadIdx.x;
  const int lane = tid & 31, wave = tid >> 5;
  const int m = lane & 15, hs = lane >> 4;
  const u16* Q  = qbf + (size_t)bh * SS * DQK;
  const u16* Kp = kbf + (size_t)bh * SS * DQK;
  const u16* Vp = vbf + (size_t)bh * SS * DV;
  const int qbase = qtile * 64 + wave * 16;
  const u32 ldsK = (u32)(uintptr_t)&sK[0];
  const u64 baseK = (u64)(uintptr_t)Kp;

  // Preload the wave's Q fragments (16 rows x 192) into registers.
  v16bf qfrag[6];
  {
    const u16* qrow = Q + (size_t)(qbase + m) * DQK;
#pragma unroll
    for (int c = 0; c < 6; ++c) {
      U4 lo = *(const U4*)(qrow + c * 32 + hs * 16);
      U4 hi = *(const U4*)(qrow + c * 32 + hs * 16 + 8);
      qfrag[c] = mkfrag(lo, hi);
    }
  }

  const v8f vzero = {0.f,0.f,0.f,0.f,0.f,0.f,0.f,0.f};
  v8f O[8];
#pragma unroll
  for (int n = 0; n < 8; ++n) O[n] = vzero;
  float mrow[8], lrow[8];
#pragma unroll
  for (int r = 0; r < 8; ++r) { mrow[r] = -3.0e38f; lrow[r] = 0.f; }

  const float SC = 0.07216878364870323f;   // 192^-0.5
  const int qmax = qtile * 64 + 63;        // uniform causal bound per block

  for (int kb = 0; kb <= qmax; kb += 32) {
    __syncthreads();   // previous iteration's readers done before overwrite
    // DMA K tile: 32 keys x 192 elems, contiguous rows
    if (wave == 0)
      tdm_load_2d(ldsK, baseK + (u64)kb * DQK * 2, DQK, 32, DQK, 0);
    // Stage V tile transposed: [col 0..127][key 0..31] (overlaps the TDM)
    for (int v = tid; v < 512; v += 128) {
      const int kk = v >> 4, c8 = (v & 15) * 8;
      U4 d = *(const U4*)(Vp + (size_t)(kb + kk) * DV + c8);
      u16 tmp[8];
      *(U4*)tmp = d;
#pragma unroll
      for (int j = 0; j < 8; ++j) sV[(c8 + j) * 40 + kk] = tmp[j];
    }
    if (wave == 0) __builtin_amdgcn_s_wait_tensorcnt(0);
    __syncthreads();

    // S = Q * K^T : two 16x16 tiles (keys kb..kb+15, kb+16..kb+31)
    v8f s0 = vzero, s1 = vzero;
#pragma unroll
    for (int c = 0; c < 6; ++c) {
      U4 l0 = *(const U4*)&sK[m * 192 + c * 32 + hs * 16];
      U4 h0 = *(const U4*)&sK[m * 192 + c * 32 + hs * 16 + 8];
      s0 = WMMA_BF16(qfrag[c], mkfrag(l0, h0), s0);
      U4 l1 = *(const U4*)&sK[(16 + m) * 192 + c * 32 + hs * 16];
      U4 h1 = *(const U4*)&sK[(16 + m) * 192 + c * 32 + hs * 16 + 8];
      s1 = WMMA_BF16(qfrag[c], mkfrag(l1, h1), s1);
    }

    // Scale, causal mask, online softmax update.
#pragma unroll
    for (int r = 0; r < 8; ++r) {
      const int row = qbase + r + 8 * hs;
      float x0 = s0[r] * SC; if (kb + m > row)      x0 = -3.0e38f;
      float x1 = s1[r] * SC; if (kb + 16 + m > row) x1 = -3.0e38f;
      float mx = fmaxf(x0, x1);
      mx = fmaxf(mx, __shfl_xor(mx, 1, 32));
      mx = fmaxf(mx, __shfl_xor(mx, 2, 32));
      mx = fmaxf(mx, __shfl_xor(mx, 4, 32));
      mx = fmaxf(mx, __shfl_xor(mx, 8, 32));
      const float mn = fmaxf(mrow[r], mx);
      const float a  = __expf(mrow[r] - mn);
      const float p0 = __expf(x0 - mn);
      const float p1 = __expf(x1 - mn);
      float rs = p0 + p1;
      rs += __shfl_xor(rs, 1, 32);
      rs += __shfl_xor(rs, 2, 32);
      rs += __shfl_xor(rs, 4, 32);
      rs += __shfl_xor(rs, 8, 32);
      lrow[r] = lrow[r] * a + rs;
      mrow[r] = mn;
      s0[r] = p0; s1[r] = p1;
#pragma unroll
      for (int n = 0; n < 8; ++n) O[n][r] *= a;
    }

    // Re-layout P (D-layout f32 -> A-layout bf16) via per-wave LDS scratch.
    u16* P = &sP[wave][0];
#pragma unroll
    for (int r = 0; r < 8; ++r) {
      P[(r + 8 * hs) * 32 + m]      = f2bf(s0[r]);
      P[(r + 8 * hs) * 32 + 16 + m] = f2bf(s1[r]);
    }
    asm volatile("s_wait_dscnt 0" ::: "memory");   // cross-lane LDS write->read
    U4 plo = *(const U4*)&P[m * 32 + hs * 16];
    U4 phi = *(const U4*)&P[m * 32 + hs * 16 + 8];
    v16bf pf = mkfrag(plo, phi);

    // O += P(16x32) * V(32x128)
#pragma unroll
    for (int n = 0; n < 8; ++n) {
      U4 vlo = *(const U4*)&sV[(n * 16 + m) * 40 + hs * 16];
      U4 vhi = *(const U4*)&sV[(n * 16 + m) * 40 + hs * 16 + 8];
      O[n] = WMMA_BF16(pf, mkfrag(vlo, vhi), O[n]);
    }
  }

  // Epilogue: normalize and store bf16 into [B*S][H*128]
#pragma unroll
  for (int r = 0; r < 8; ++r) {
    const float inv = 1.0f / lrow[r];
    const int srow = qbase + r + 8 * hs;
    u16* dst = attn + (size_t)(b * SS + srow) * (HH * DV) + head * DV;
#pragma unroll
    for (int n = 0; n < 8; ++n) dst[n * 16 + m] = f2bf(O[n][r] * inv);
  }
}

// ---------------------------------------------------------------------------
// Orchestration
// ---------------------------------------------------------------------------
extern "C" void kernel_launch(void* const* d_in, const int* in_sizes, int n_in,
                              void* d_out, int out_size, void* d_ws, size_t ws_size,
                              hipStream_t stream) {
  (void)in_sizes; (void)n_in; (void)out_size; (void)ws_size;
  const float* x     = (const float*)d_in[0];
  const float* fc    = (const float*)d_in[1];
  const float* wq_a  = (const float*)d_in[2];
  const float* wq_b  = (const float*)d_in[3];
  const float* q_nw  = (const float*)d_in[4];
  const float* wkv_a = (const float*)d_in[5];
  const float* kv_nw = (const float*)d_in[6];
  const float* wkv_b = (const float*)d_in[7];
  const float* wo    = (const float*)d_in[8];
  float* out = (float*)d_out;

  char* p = (char*)d_ws;
  auto alloc = [&](size_t bytes) -> void* {
    void* r = (void*)p;
    p += (bytes + 255) & ~(size_t)255;
    return r;
  };
  u16*   x_bf     = (u16*)  alloc((size_t)BSROWS * DIMM * 2);
  u16*   wqa_bf   = (u16*)  alloc((size_t)512 * DIMM * 2);
  u16*   wqb_bf   = (u16*)  alloc((size_t)(HH * DQK) * 512 * 2);
  u16*   wkva_bf  = (u16*)  alloc((size_t)576 * DIMM * 2);
  u16*   wkvb_bf  = (u16*)  alloc((size_t)(HH * 256) * 512 * 2);
  u16*   wo_bf    = (u16*)  alloc((size_t)DIMM * (HH * DV) * 2);
  float* qlat     = (float*)alloc((size_t)BSROWS * 512 * 4);
  u16*   qlat_bf  = (u16*)  alloc((size_t)BSROWS * 512 * 2);
  float* qfull    = (float*)alloc((size_t)BSROWS * HH * DQK * 4);
  float* kva      = (float*)alloc((size_t)BSROWS * 576 * 4);
  u16*   kvlat_bf = (u16*)  alloc((size_t)BSROWS * 512 * 2);
  float* kvb      = (float*)alloc((size_t)BSROWS * HH * 256 * 4);
  u16*   q_bf     = (u16*)  alloc((size_t)BB * HH * SS * DQK * 2);
  u16*   k_bf     = (u16*)  alloc((size_t)BB * HH * SS * DQK * 2);
  u16*   v_bf     = (u16*)  alloc((size_t)BB * HH * SS * DV * 2);
  u16*   attn_bf  = (u16*)qfull;   // reuse: qfull dead after rope_q

  const dim3 blk256(256);
  auto cvt = [&](const float* src, u16* dst, size_t n) {
    cvt_f32_bf16<<<dim3(2048), blk256, 0, stream>>>(src, dst, (int)n);
  };
  cvt(x,     x_bf,    (size_t)BSROWS * DIMM);
  cvt(wq_a,  wqa_bf,  (size_t)512 * DIMM);
  cvt(wq_b,  wqb_bf,  (size_t)HH * DQK * 512);
  cvt(wkv_a, wkva_bf, (size_t)576 * DIMM);
  cvt(wkv_b, wkvb_bf, (size_t)HH * 256 * 512);
  cvt(wo,    wo_bf,   (size_t)DIMM * HH * DV);

  gemm_bf16_nt<<<dim3(512 / 64, BSROWS / 64), blk256, 0, stream>>>(
      x_bf, wqa_bf, qlat, BSROWS, 512, DIMM);
  gemm_bf16_nt<<<dim3(576 / 64, BSROWS / 64), blk256, 0, stream>>>(
      x_bf, wkva_bf, kva, BSROWS, 576, DIMM);
  rmsnorm_bf16<<<dim3(BSROWS), blk256, 0, stream>>>(qlat, q_nw, qlat_bf, 512, 512);
  rmsnorm_bf16<<<dim3(BSROWS), blk256, 0, stream>>>(kva, kv_nw, kvlat_bf, 576, 512);
  gemm_bf16_nt<<<dim3((HH * DQK) / 64, BSROWS / 64), blk256, 0, stream>>>(
      qlat_bf, wqb_bf, qfull, BSROWS, HH * DQK, 512);
  gemm_bf16_nt<<<dim3((HH * 256) / 64, BSROWS / 64), blk256, 0, stream>>>(
      kvlat_bf, wkvb_bf, kvb, BSROWS, HH * 256, 512);
  rope_q_kernel<<<dim3(BSROWS * HH), dim3(96), 0, stream>>>(qfull, fc, q_bf);
  assemble_kv_kernel<<<dim3(BSROWS * HH), dim3(96), 0, stream>>>(kvb, kva, fc, k_bf, v_bf);
  mla_flash_attn<<<dim3(SS / 64, BB * HH), dim3(128), 0, stream>>>(q_bf, k_bf, v_bf, attn_bf);
  gemm_bf16_nt<<<dim3(DIMM / 64, BSROWS / 64), blk256, 0, stream>>>(
      attn_bf, wo_bf, out, BSROWS, DIMM, HH * DV);
}
